// HybridModel_29137058136266
// MI455X (gfx1250) — matrix operand order repaired and checked by reading, not compiled, not run on hardware
//
#include <hip/hip_runtime.h>
#include <hip/hip_bf16.h>

// CDNA5 / gfx1250 WMMA types
typedef __attribute__((ext_vector_type(16))) _Float16 v16h;
typedef __attribute__((ext_vector_type(8)))  _Float16 v8h;
typedef __attribute__((ext_vector_type(8)))  float    v8f;
typedef __attribute__((ext_vector_type(4)))  float    v4f;

#define N_NODES 100000
#define N_EDGES 3200000

// ---------------------------------------------------------------------------
// Fragment layouts (ISA 7.12.2, wave32):
//  A (16-bit 16xK): lanes 0-15: elements 0..7 -> K 0..7, 8..15 -> K 16..23
//                   lanes 16-31: elements 0..7 -> K 8..15, 8..15 -> K 24..31
//                   => per lane: two contiguous 8-element K runs at kb, kb+16
//  B (Kx16):        lanes 0-15: K 0..7,16..23 (col N=lane); lanes 16-31: K 8..15,24..31
//                   => per lane: two contiguous 8-element K runs at kbB, kbB+16
//  C/D (16x16 f32): lanes 0-15: N=lane, VGPR r -> M=r; lanes 16-31: M=8+r
// All runs are loaded with 128-bit vector loads (alignment guaranteed by the
// 8/16-element offsets and 256B-aligned workspace buffers).
// ---------------------------------------------------------------------------

// K1: fused conv1(1->16,k3,pad1)+relu + conv2(16->32,k3,pad1)+relu + mean.
// One wave per node; conv2 = 32x48 weights @ 48x64 im2col as 2x4 WMMA tiles
// over two K-steps (K padded 48->64).
__global__ __launch_bounds__(256)
void conv_fused_kernel(const float* __restrict__ x,
                       const float* __restrict__ w1, const float* __restrict__ b1,
                       const float* __restrict__ w2, const float* __restrict__ b2,
                       _Float16* __restrict__ h16)            // out: [N,32] f16
{
    __shared__ float    lx[8][66];         // padded input row per wave
    __shared__ _Float16 lh1[8][16][66];    // padded conv1 activations per wave

    const int wave = threadIdx.x >> 5;
    const int lane = threadIdx.x & 31;
    const int node = blockIdx.x * 8 + wave;   // 12500*8 == 100000 exactly

    // stage x[node] with zero pad
    lx[wave][1 + lane]  = x[node * 64 + lane];
    lx[wave][33 + lane] = x[node * 64 + 32 + lane];
    if (lane == 0) { lx[wave][0] = 0.f; lx[wave][65] = 0.f; }
    __syncthreads();

    // conv1 + relu -> lh1 (f16, padded). lane -> (channel, 32 positions)
    {
        const int c  = lane & 15;
        const int hp = (lane >> 4) * 32;
        const float wA = w1[c * 3 + 0], wB = w1[c * 3 + 1], wC = w1[c * 3 + 2];
        const float bb = b1[c];
        if (lane < 16) { lh1[wave][c][0] = (_Float16)0.f; lh1[wave][c][65] = (_Float16)0.f; }
        #pragma unroll 4
        for (int p = 0; p < 32; ++p) {
            const int q = hp + p;
            float v = wA * lx[wave][q] + wB * lx[wave][q + 1] + wC * lx[wave][q + 2] + bb;
            v = v > 0.f ? v : 0.f;
            lh1[wave][c][q + 1] = (_Float16)v;
        }
    }
    __syncthreads();

    // conv2 weight fragments A[c2tile][kstep] via 128-bit loads; K = c1*3+k
    const int mb = (lane < 16) ? 0 : 8;
    v16h Af[2][2];
    #pragma unroll
    for (int ct = 0; ct < 2; ++ct)
        #pragma unroll
        for (int ks = 0; ks < 2; ++ks) {
            const int c2 = ct * 16 + (lane & 15);
            const float* wp = w2 + c2 * 48 + ks * 32 + mb;
            const v4f q0 = *(const v4f*)(wp);
            const v4f q1 = *(const v4f*)(wp + 4);
            v4f q2 = {0.f, 0.f, 0.f, 0.f}, q3 = {0.f, 0.f, 0.f, 0.f};
            if (ks == 0) {              // second run (K+16) only valid for ks=0
                q2 = *(const v4f*)(wp + 16);
                q3 = *(const v4f*)(wp + 20);
            }
            #pragma unroll
            for (int e = 0; e < 4; ++e) {
                Af[ct][ks][e]      = (_Float16)q0[e];
                Af[ct][ks][e + 4]  = (_Float16)q1[e];
                Af[ct][ks][e + 8]  = (_Float16)q2[e];
                Af[ct][ks][e + 12] = (_Float16)q3[e];
            }
        }

    const int col   = lane & 15;
    const int rbase = (lane < 16) ? 0 : 8;
    const v4f bq0 = *(const v4f*)(b2 + rbase);
    const v4f bq1 = *(const v4f*)(b2 + rbase + 4);
    const v4f bq2 = *(const v4f*)(b2 + 16 + rbase);
    const v4f bq3 = *(const v4f*)(b2 + 20 + rbase);
    float sum0[8], sum1[8], b2v0[8], b2v1[8];
    #pragma unroll
    for (int r = 0; r < 4; ++r) {
        b2v0[r] = bq0[r]; b2v0[r + 4] = bq1[r];
        b2v1[r] = bq2[r]; b2v1[r + 4] = bq3[r];
    }
    #pragma unroll
    for (int r = 0; r < 8; ++r) { sum0[r] = 0.f; sum1[r] = 0.f; }

    for (int pt = 0; pt < 4; ++pt) {            // four 16-wide position tiles
        const int p = pt * 16 + col;
        v16h Bf[2];
        const int kbB = (lane < 16) ? 0 : 8;
        #pragma unroll
        for (int ks = 0; ks < 2; ++ks) {
            #pragma unroll
            for (int e = 0; e < 16; ++e) {
                const int K = ks * 32 + kbB + e + (e >= 8 ? 8 : 0);
                _Float16 v = (_Float16)0.f;
                if (K < 48) {                    // im2col from LDS (scattered)
                    const int c1 = K / 3, kk = K - c1 * 3;
                    v = lh1[wave][c1][p + kk];
                }
                Bf[ks][e] = v;
            }
        }
        v8f acc0 = {}, acc1 = {};
        acc0 = __builtin_amdgcn_wmma_f32_16x16x32_f16(false, Af[0][0], false, Bf[0], (short)0, acc0, false, false);
        acc0 = __builtin_amdgcn_wmma_f32_16x16x32_f16(false, Af[0][1], false, Bf[1], (short)0, acc0, false, false);
        acc1 = __builtin_amdgcn_wmma_f32_16x16x32_f16(false, Af[1][0], false, Bf[0], (short)0, acc1, false, false);
        acc1 = __builtin_amdgcn_wmma_f32_16x16x32_f16(false, Af[1][1], false, Bf[1], (short)0, acc1, false, false);
        #pragma unroll
        for (int r = 0; r < 8; ++r) {           // bias + relu + position-sum
            float v0 = acc0[r] + b2v0[r]; sum0[r] += (v0 > 0.f ? v0 : 0.f);
            float v1 = acc1[r] + b2v1[r]; sum1[r] += (v1 > 0.f ? v1 : 0.f);
        }
    }

    // sum over the 16 position-lanes of each half-wave
    #pragma unroll
    for (int off = 1; off < 16; off <<= 1) {
        #pragma unroll
        for (int r = 0; r < 8; ++r) {
            sum0[r] += __shfl_xor(sum0[r], off, 32);
            sum1[r] += __shfl_xor(sum1[r], off, 32);
        }
    }
    if ((lane & 15) == 0) {
        v8h o0, o1;
        #pragma unroll
        for (int r = 0; r < 8; ++r) {
            o0[r] = (_Float16)(sum0[r] * (1.f / 64.f));
            o1[r] = (_Float16)(sum1[r] * (1.f / 64.f));
        }
        *(v8h*)(h16 + node * 32 + rbase)      = o0;   // 128-bit stores
        *(v8h*)(h16 + node * 32 + 16 + rbase) = o1;
    }
}

// K2: in-degree counts (float, feeds mean divisor)
__global__ __launch_bounds__(256)
void degree_kernel(const long long* __restrict__ dst, float* __restrict__ cnt)
{
    const int i = blockIdx.x * blockDim.x + threadIdx.x;
    if (i < N_EDGES) atomicAdd(&cnt[(int)dst[i]], 1.0f);
}

// K3: edge scatter. One wave = one 32-feature slice of one edge; edge index is
// wave-uniform (readfirstlane -> scalar load). f16 gather (L2-resident source),
// f32 atomic accumulate (global_atomic_add_f32 in L2).
template <int SHIFT>
__global__ __launch_bounds__(256)
void scatter_kernel(const long long* __restrict__ src,
                    const long long* __restrict__ dst,
                    const _Float16* __restrict__ hsrc,   // [N, F]
                    float* __restrict__ agg, int F)
{
    const long long i = (long long)blockIdx.x * blockDim.x + threadIdx.x;
    if (i >= (long long)N_EDGES << SHIFT) return;
    const int e = __builtin_amdgcn_readfirstlane((int)(i >> SHIFT));
    const int f = (int)(i & ((1 << SHIFT) - 1));
    const int s = (int)src[e];
    const int d = (int)dst[e];
    const float v = (float)hsrc[s * F + f];
    atomicAdd(&agg[d * F + f], v);
}

// K4: SAGE1 combine: relu(mean(agg) @ lw^T + lb + h @ rw^T) -> f16 [N,64]
// One wave per 16-node tile; K=32 (one WMMA step), 4 output tiles of 16.
__global__ __launch_bounds__(256)
void sage1_kernel(const float* __restrict__ agg, const float* __restrict__ cnt,
                  const _Float16* __restrict__ hself,      // [N,32] f16
                  const float* __restrict__ lw, const float* __restrict__ lb,
                  const float* __restrict__ rw,
                  _Float16* __restrict__ out)              // [N,64] f16
{
    const int wave = threadIdx.x >> 5;
    const int lane = threadIdx.x & 31;
    const int tile = blockIdx.x * 8 + wave;
    if (tile >= N_NODES / 16) return;
    const int n0   = tile * 16;
    const int row  = lane & 15;
    const int node = n0 + row;

    const float ic  = 1.f / fmaxf(cnt[node], 1.f);
    const int   kbA = (lane < 16) ? 0 : 8;

    // A fragments: two contiguous 8-K runs per lane -> 128-bit loads
    const float* ap = agg + node * 32 + kbA;
    const v4f a0 = *(const v4f*)(ap);
    const v4f a1 = *(const v4f*)(ap + 4);
    const v4f a2 = *(const v4f*)(ap + 16);
    const v4f a3 = *(const v4f*)(ap + 20);
    const v8h s0 = *(const v8h*)(hself + node * 32 + kbA);
    const v8h s1 = *(const v8h*)(hself + node * 32 + kbA + 16);
    v16h A1, A2;
    #pragma unroll
    for (int e = 0; e < 4; ++e) {
        A1[e]      = (_Float16)(a0[e] * ic);
        A1[e + 4]  = (_Float16)(a1[e] * ic);
        A1[e + 8]  = (_Float16)(a2[e] * ic);
        A1[e + 12] = (_Float16)(a3[e] * ic);
    }
    #pragma unroll
    for (int e = 0; e < 8; ++e) { A2[e] = s0[e]; A2[e + 8] = s1[e]; }

    const int kbB   = (lane < 16) ? 0 : 8;
    const int rbase = (lane < 16) ? 0 : 8;
    for (int ot = 0; ot < 4; ++ot) {
        const int o = ot * 16 + row;
        // B fragments: two contiguous 8-K runs per lane at kbB, kbB+16 -> 4x float4 each
        const float* lp = lw + o * 32 + kbB;
        const float* rp = rw + o * 32 + kbB;
        v16h B1, B2;
        #pragma unroll
        for (int c = 0; c < 4; ++c) {
            const v4f lv = *(const v4f*)(lp + 4 * c + (c & 2) * 4);
            const v4f rv = *(const v4f*)(rp + 4 * c + (c & 2) * 4);
            #pragma unroll
            for (int e = 0; e < 4; ++e) {
                B1[4 * c + e] = (_Float16)lv[e];
                B2[4 * c + e] = (_Float16)rv[e];
            }
        }
        v8f acc = {};
        acc = __builtin_amdgcn_wmma_f32_16x16x32_f16(false, A1, false, B1, (short)0, acc, false, false);
        acc = __builtin_amdgcn_wmma_f32_16x16x32_f16(false, A2, false, B2, (short)0, acc, false, false);
        const float bias = lb[o];
        #pragma unroll
        for (int r = 0; r < 8; ++r) {
            float v = acc[r] + bias;
            v = v > 0.f ? v : 0.f;
            out[(n0 + rbase + r) * 64 + o] = (_Float16)v;
        }
    }
}

// K5: SAGE2 combine (K=64 -> 2 WMMA K-steps per GEMM) + head dot, fused.
__global__ __launch_bounds__(256)
void sage2_head_kernel(const float* __restrict__ agg, const float* __restrict__ cnt,
                       const _Float16* __restrict__ hself,   // [N,64] f16
                       const float* __restrict__ lw, const float* __restrict__ lb,
                       const float* __restrict__ rw,
                       const float* __restrict__ hw, const float* __restrict__ hb,
                       float* __restrict__ out)              // [N] f32
{
    const int wave = threadIdx.x >> 5;
    const int lane = threadIdx.x & 31;
    const int tile = blockIdx.x * 8 + wave;
    if (tile >= N_NODES / 16) return;
    const int n0   = tile * 16;
    const int row  = lane & 15;
    const int node = n0 + row;

    const float ic  = 1.f / fmaxf(cnt[node], 1.f);
    const int   kbA = (lane < 16) ? 0 : 8;

    v16h A1[2], A2[2];
    #pragma unroll
    for (int ks = 0; ks < 2; ++ks) {
        const float* ap = agg + node * 64 + ks * 32 + kbA;
        const v4f a0 = *(const v4f*)(ap);
        const v4f a1 = *(const v4f*)(ap + 4);
        const v4f a2 = *(const v4f*)(ap + 16);
        const v4f a3 = *(const v4f*)(ap + 20);
        const v8h s0 = *(const v8h*)(hself + node * 64 + ks * 32 + kbA);
        const v8h s1 = *(const v8h*)(hself + node * 64 + ks * 32 + kbA + 16);
        #pragma unroll
        for (int e = 0; e < 4; ++e) {
            A1[ks][e]      = (_Float16)(a0[e] * ic);
            A1[ks][e + 4]  = (_Float16)(a1[e] * ic);
            A1[ks][e + 8]  = (_Float16)(a2[e] * ic);
            A1[ks][e + 12] = (_Float16)(a3[e] * ic);
        }
        #pragma unroll
        for (int e = 0; e < 8; ++e) { A2[ks][e] = s0[e]; A2[ks][e + 8] = s1[e]; }
    }

    const int kbB   = (lane < 16) ? 0 : 8;
    const int rbase = (lane < 16) ? 0 : 8;
    float part[8];
    #pragma unroll
    for (int r = 0; r < 8; ++r) part[r] = 0.f;

    for (int ot = 0; ot < 4; ++ot) {
        const int o = ot * 16 + row;
        v16h B1[2], B2[2];
        #pragma unroll
        for (int ks = 0; ks < 2; ++ks) {
            const float* lp = lw + o * 64 + ks * 32 + kbB;
            const float* rp = rw + o * 64 + ks * 32 + kbB;
            #pragma unroll
            for (int c = 0; c < 4; ++c) {
                const v4f lv = *(const v4f*)(lp + 4 * c + (c & 2) * 4);
                const v4f rv = *(const v4f*)(rp + 4 * c + (c & 2) * 4);
                #pragma unroll
                for (int e = 0; e < 4; ++e) {
                    B1[ks][4 * c + e] = (_Float16)lv[e];
                    B2[ks][4 * c + e] = (_Float16)rv[e];
                }
            }
        }
        v8f acc = {};
        acc = __builtin_amdgcn_wmma_f32_16x16x32_f16(false, A1[0], false, B1[0], (short)0, acc, false, false);
        acc = __builtin_amdgcn_wmma_f32_16x16x32_f16(false, A1[1], false, B1[1], (short)0, acc, false, false);
        acc = __builtin_amdgcn_wmma_f32_16x16x32_f16(false, A2[0], false, B2[0], (short)0, acc, false, false);
        acc = __builtin_amdgcn_wmma_f32_16x16x32_f16(false, A2[1], false, B2[1], (short)0, acc, false, false);
        const float bias = lb[o];
        const float w    = hw[o];
        #pragma unroll
        for (int r = 0; r < 8; ++r) part[r] += (acc[r] + bias) * w;
    }

    // reduce over the 16 feature-lanes of each half-wave
    #pragma unroll
    for (int off = 1; off < 16; off <<= 1)
        #pragma unroll
        for (int r = 0; r < 8; ++r) part[r] += __shfl_xor(part[r], off, 32);

    if ((lane & 15) == 0) {
        const float hbv = hb[0];
        #pragma unroll
        for (int r = 0; r < 8; ++r) out[n0 + rbase + r] = part[r] + hbv;
    }
}

extern "C" void kernel_launch(void* const* d_in, const int* in_sizes, int n_in,
                              void* d_out, int out_size, void* d_ws, size_t ws_size,
                              hipStream_t stream)
{
    const float*     x    = (const float*)d_in[0];
    const long long* eidx = (const long long*)d_in[1];
    const float* w1   = (const float*)d_in[2];
    const float* b1   = (const float*)d_in[3];
    const float* w2   = (const float*)d_in[4];
    const float* b2   = (const float*)d_in[5];
    const float* s1lw = (const float*)d_in[6];
    const float* s1lb = (const float*)d_in[7];
    const float* s1rw = (const float*)d_in[8];
    const float* s2lw = (const float*)d_in[9];
    const float* s2lb = (const float*)d_in[10];
    const float* s2rw = (const float*)d_in[11];
    const float* hw   = (const float*)d_in[12];
    const float* hb   = (const float*)d_in[13];

    char*  ws  = (char*)d_ws;
    size_t off = 0;
    auto alloc = [&](size_t bytes) -> void* {
        void* p = ws + off;
        off = (off + bytes + 255) & ~(size_t)255;
        return p;
    };
    _Float16* h16  = (_Float16*)alloc((size_t)N_NODES * 32 * 2);  // 6.4 MB
    float*    agg1 = (float*)   alloc((size_t)N_NODES * 32 * 4);  // 12.8 MB
    float*    cnt  = (float*)   alloc((size_t)N_NODES * 4);       // 0.4 MB
    _Float16* h64  = (_Float16*)alloc((size_t)N_NODES * 64 * 2);  // 12.8 MB
    float*    agg2 = (float*)   alloc((size_t)N_NODES * 64 * 4);  // 25.6 MB

    const long long* esrc = eidx;
    const long long* edst = eidx + N_EDGES;

    hipMemsetAsync(agg1, 0, (size_t)N_NODES * 32 * 4, stream);
    hipMemsetAsync(cnt,  0, (size_t)N_NODES * 4,      stream);
    hipMemsetAsync(agg2, 0, (size_t)N_NODES * 64 * 4, stream);

    conv_fused_kernel<<<N_NODES / 8, 256, 0, stream>>>(x, w1, b1, w2, b2, h16);

    degree_kernel<<<(N_EDGES + 255) / 256, 256, 0, stream>>>(edst, cnt);

    scatter_kernel<5><<<(int)(((long long)N_EDGES * 32 + 255) / 256), 256, 0, stream>>>(
        esrc, edst, h16, agg1, 32);

    sage1_kernel<<<(N_NODES / 16 + 7) / 8, 256, 0, stream>>>(
        agg1, cnt, h16, s1lw, s1lb, s1rw, h64);

    scatter_kernel<6><<<(int)(((long long)N_EDGES * 64 + 255) / 256), 256, 0, stream>>>(
        esrc, edst, h64, agg2, 64);

    sage2_head_kernel<<<(N_NODES / 16 + 7) / 8, 256, 0, stream>>>(
        agg2, cnt, h64, s2lw, s2lb, s2rw, hw, hb, (float*)d_out);
}
